// EnergyHead_H_78262894068053
// MI455X (gfx1250) — matrix operand order repaired and checked by reading, not compiled, not run on hardware
//
#include <hip/hip_runtime.h>

// ============================================================================
// EnergyHead: out = -( softmax(causal(x @ (xW^T)^T)) @ (xW^T) )
// B=8, T=2048, C=1024, fp32 in/out. Strategy: f16 WMMA (16x16x32) with f32
// accumulation; flash-style two-pass softmax (stats pass + output pass).
// f16 tile staging uses CDNA5 async global->LDS DMA (ASYNCcnt).
// ============================================================================

constexpr int Bn = 8;
constexpr int Tn = 2048;
constexpr int Cn = 1024;

typedef __attribute__((ext_vector_type(16))) _Float16 v16h;
typedef __attribute__((ext_vector_type(8)))  _Float16 v8h;
typedef __attribute__((ext_vector_type(8)))  float    v8f;

union FragU { v16h v; v8h h[2]; };

constexpr int TP = 40;  // LDS pitch (halves) for 64x32 tiles (32 + 8 pad)
constexpr int PP = 72;  // LDS pitch (halves) for 64x64 P tile (64 + 8 pad)

// A-operand fragment, 16x32 f16.  Per CDNA5 ISA 7.12.2: lanes 0-15 (row M)
// hold K=0..7 (v0-3) and K=16..23 (v4-7); lanes 16-31 hold K=8..15 / 24..31.
__device__ __forceinline__ v16h ldsFragA(const _Float16* base, int row0,
                                         int pitch, int klo, int mrow, int hi) {
  const _Float16* p = base + (row0 + mrow) * pitch + klo + hi * 8;
  FragU u;
  u.h[0] = *(const v8h*)(p);        // K = klo + hi*8 + 0..7
  u.h[1] = *(const v8h*)(p + 16);   // K = klo + 16 + hi*8 + 0..7
  return u.v;
}

// B-operand fragment, 32x16 f16.  Lanes 0-15 hold K=0..7 / 16..23 for column
// N=lane, lanes 16-31 hold K=8..15 / 24..31 (matches A fragment K order).
__device__ __forceinline__ v16h ldsFragB(const _Float16* base, int col0,
                                         int pitch, int klo, int ncol, int hi) {
  const _Float16* p = base + (col0 + ncol) * pitch + klo + hi * 8;
  FragU u;
  u.h[0] = *(const v8h*)(p);
  u.h[1] = *(const v8h*)(p + 16);
  return u.v;
}

__device__ __forceinline__ v8f wmma_f16(v16h a, v16h b, v8f c) {
  return __builtin_amdgcn_wmma_f32_16x16x32_f16(false, a, false, b,
                                                (short)0, c, false, false);
}

// ---- CDNA5 async global->LDS DMA (GV mode, per-lane 128-bit transfer) ----
// Low 32 bits of a flat pointer into LDS are the wave-relative LDS address
// (ISA 10.2); that is what VDST of the async op wants.
__device__ __forceinline__ void asyncLoadB128(_Float16* ldst,
                                              const _Float16* gsrc) {
  const uint32_t loff = (uint32_t)(uintptr_t)ldst;
  asm volatile("global_load_async_to_lds_b128 %0, %1, off"
               :
               : "v"(loff), "v"(gsrc)
               : "memory");
}
__device__ __forceinline__ void asyncWait0() {
  asm volatile("s_wait_asynccnt 0x0" ::: "memory");
}

// Stage a 64x32 f32 tile from global into LDS, converting to f16. 256 threads.
__device__ __forceinline__ void stage_cvt64x32(_Float16* dst, const float* src,
                                               int srcPitch, int tid) {
  int q = tid;
#pragma unroll
  for (int it = 0; it < 2; ++it, q += 256) {
    const int r = q >> 3, c4 = (q & 7) << 2;
    const float4 f = *(const float4*)(src + (size_t)r * srcPitch + c4);
    _Float16* d = dst + r * TP + c4;
    d[0] = (_Float16)f.x; d[1] = (_Float16)f.y;
    d[2] = (_Float16)f.z; d[3] = (_Float16)f.w;
  }
}

// Stage a 64x32 f16 tile global->LDS via async DMA. 256 threads, 16B each.
// Caller must asyncWait0() + __syncthreads() before consuming.
__device__ __forceinline__ void stage_h64x32_async(_Float16* dst,
                                                   const _Float16* src,
                                                   int srcPitch, int tid) {
  const int r = tid >> 2, c8 = (tid & 3) << 3;
  asyncLoadB128(dst + r * TP + c8, src + (size_t)r * srcPitch + c8);
}

// ============================================================================
// Kernel 1: xh[b,t,d] = sum_c x[b,t,c] * W[d,c]   (f16 out, plus transposed
// copy xhT[b,d,t] so the later P@xH GEMM has a contiguous B operand).
// Block: 256 threads (8 waves), 64x64 output tile; wave -> 16x32 sub-tile.
// ============================================================================
__global__ __launch_bounds__(256) void k_gemm_xh(const float* __restrict__ x,
                                                 const float* __restrict__ W,
                                                 _Float16* __restrict__ xh,
                                                 _Float16* __restrict__ xhT) {
  __shared__ __align__(16) _Float16 As[64 * TP];
  __shared__ __align__(16) _Float16 Bs[64 * TP];
  const int tBase = blockIdx.x * 64;
  const int dBase = blockIdx.y * 64;
  const int b = blockIdx.z;
  const int tid = threadIdx.x;
  const int w = tid >> 5, lane = tid & 31, mrow = lane & 15, hi = lane >> 4;
  const int ra = (w & 3) << 4;   // wave's row-tile base within 64
  const int ca = (w >> 2) << 5;  // wave's 32-wide col chunk base
  const float* xb = x + (size_t)b * Tn * Cn + (size_t)tBase * Cn;
  v8f acc0 = {}, acc1 = {};
  for (int kk = 0; kk < Cn; kk += 32) {
    if (kk + 32 < Cn) __builtin_prefetch(xb + kk + 32, 0, 1);  // global_prefetch_b8
    __syncthreads();
    stage_cvt64x32(As, xb + kk, Cn, tid);                        // x rows (M=t)
    stage_cvt64x32(Bs, W + (size_t)dBase * Cn + kk, Cn, tid);    // W rows (N=d)
    __syncthreads();
    const v16h a  = ldsFragA(As, ra, TP, 0, mrow, hi);
    const v16h b0 = ldsFragB(Bs, ca,      TP, 0, mrow, hi);
    const v16h b1 = ldsFragB(Bs, ca + 16, TP, 0, mrow, hi);
    acc0 = wmma_f16(a, b0, acc0);
    acc1 = wmma_f16(a, b1, acc1);
  }
#pragma unroll
  for (int v = 0; v < 8; ++v) {
    const int t = tBase + ra + v + 8 * hi;  // C/D layout: M = vgpr + 8*hi
    const int d = dBase + ca + mrow;        // N = lane%16
    const _Float16 e0 = (_Float16)acc0[v];
    const _Float16 e1 = (_Float16)acc1[v];
    xh [((size_t)b * Tn + t) * Cn + d]      = e0;
    xh [((size_t)b * Tn + t) * Cn + d + 16] = e1;
    xhT[((size_t)b * Cn + d) * Tn + t]      = e0;
    xhT[((size_t)b * Cn + d + 16) * Tn + t] = e1;
  }
}

// ============================================================================
// Kernel 2a: per-row softmax stats (online max m and denom l) over the
// strictly-causal score matrix S[t,s] = x[t,:] . xh[s,:].  Row t=0 -> l=0.
// ============================================================================
__global__ __launch_bounds__(256) void k_softmax_stats(
    const float* __restrict__ x, const _Float16* __restrict__ xh,
    float* __restrict__ mbuf, float* __restrict__ lbuf) {
  __shared__ __align__(16) _Float16 As[64 * TP];
  __shared__ __align__(16) _Float16 Bs[64 * TP];
  __shared__ float Sf[64][65];
  const int ib = blockIdx.x, b = blockIdx.y;
  const int tBase = ib * 64;
  const int tid = threadIdx.x;
  const int w = tid >> 5, lane = tid & 31, mrow = lane & 15, hi = lane >> 4;
  const int ra = (w & 3) << 4;
  const int ca = (w >> 2) << 5;
  const float* xb = x + (size_t)b * Tn * Cn + (size_t)tBase * Cn;
  const _Float16* xhb = xh + (size_t)b * Tn * Cn;
  float mrun = -3.0e38f, lrun = 0.0f;  // thread tid<64 owns row tid
  for (int j = 0; j <= ib; ++j) {
    const int sBase = j * 64;
    v8f s0 = {}, s1 = {};
    for (int kk = 0; kk < Cn; kk += 32) {
      __syncthreads();
      stage_h64x32_async(Bs, xhb + (size_t)sBase * Cn + kk, Cn, tid);  // N = s
      stage_cvt64x32(As, xb + kk, Cn, tid);
      asyncWait0();
      __syncthreads();
      const v16h a  = ldsFragA(As, ra, TP, 0, mrow, hi);
      const v16h b0 = ldsFragB(Bs, ca,      TP, 0, mrow, hi);
      const v16h b1 = ldsFragB(Bs, ca + 16, TP, 0, mrow, hi);
      s0 = wmma_f16(a, b0, s0);
      s1 = wmma_f16(a, b1, s1);
    }
#pragma unroll
    for (int v = 0; v < 8; ++v) {
      Sf[ra + v + 8 * hi][ca + mrow]      = s0[v];
      Sf[ra + v + 8 * hi][ca + mrow + 16] = s1[v];
    }
    __syncthreads();
    if (tid < 64) {  // online softmax update; masked (s>=t) entries skipped
      const int lim = (j == ib) ? tid : 64;
      if (lim > 0) {
        float bm = -3.0e38f;
        for (int sc = 0; sc < lim; ++sc) bm = fmaxf(bm, Sf[tid][sc]);
        const float mn = fmaxf(mrun, bm);
        float s = lrun * __expf(mrun - mn);
        for (int sc = 0; sc < lim; ++sc) s += __expf(Sf[tid][sc] - mn);
        mrun = mn; lrun = s;
      }
    }
  }
  if (tid < 64) {
    mbuf[(size_t)b * Tn + tBase + tid] = mrun;
    lbuf[(size_t)b * Tn + tBase + tid] = lrun;  // 0 for fully-masked row 0
  }
}

// ============================================================================
// Kernel 2b: out[b, tTile, :] = -sum_s P[t,s] * xh[s,:], recomputing score
// tiles with WMMA, P = exp(S-m)/l (strict s<t).  Wave owns a 64x128 slice of
// the 64x1024 accumulator (4x8 tiles of 16x16 = 256 acc VGPRs).
// ============================================================================
__global__ __launch_bounds__(256) void k_attn_out(
    const float* __restrict__ x, const _Float16* __restrict__ xh,
    const _Float16* __restrict__ xhT, const float* __restrict__ mbuf,
    const float* __restrict__ lbuf, float* __restrict__ out) {
  __shared__ __align__(16) _Float16 As[64 * TP];
  __shared__ __align__(16) _Float16 Bs[64 * TP];
  __shared__ __align__(16) _Float16 Pl[64 * PP];
  __shared__ __align__(16) _Float16 XT[1024 * TP];  // xh^T tile: [d][s-chunk 32]
  __shared__ float mL[64];
  __shared__ float liL[64];
  const int ib = blockIdx.x, b = blockIdx.y;
  const int tBase = ib * 64;
  const int tid = threadIdx.x;
  const int w = tid >> 5, lane = tid & 31, mrow = lane & 15, hi = lane >> 4;
  const int ra = (w & 3) << 4;
  const int ca = (w >> 2) << 5;
  const float* xb = x + (size_t)b * Tn * Cn + (size_t)tBase * Cn;
  const _Float16* xhb  = xh  + (size_t)b * Tn * Cn;
  const _Float16* xhTb = xhT + (size_t)b * Cn * Tn;
  if (tid < 64) {
    const float m = mbuf[(size_t)b * Tn + tBase + tid];
    const float l = lbuf[(size_t)b * Tn + tBase + tid];
    mL[tid] = m;
    liL[tid] = (l > 0.0f) ? (1.0f / l) : 0.0f;  // l==0 only for global row 0
  }
  v8f acc[4][8] = {};
  for (int j = 0; j <= ib; ++j) {
    const int sBase = j * 64;
    // ---- Phase A: recompute 64x64 score tile ----
    v8f s0 = {}, s1 = {};
    for (int kk = 0; kk < Cn; kk += 32) {
      __syncthreads();
      stage_h64x32_async(Bs, xhb + (size_t)sBase * Cn + kk, Cn, tid);
      stage_cvt64x32(As, xb + kk, Cn, tid);
      asyncWait0();
      __syncthreads();
      const v16h a  = ldsFragA(As, ra, TP, 0, mrow, hi);
      const v16h b0 = ldsFragB(Bs, ca,      TP, 0, mrow, hi);
      const v16h b1 = ldsFragB(Bs, ca + 16, TP, 0, mrow, hi);
      s0 = wmma_f16(a, b0, s0);
      s1 = wmma_f16(a, b1, s1);
    }
    // ---- P = exp(S - m) / l with strict causal mask, f16 into LDS ----
#pragma unroll
    for (int v = 0; v < 8; ++v) {
      const int row = ra + v + 8 * hi;
      const int tg  = tBase + row;
      const int sg0 = sBase + ca + mrow;
      const float p0 = (sg0      < tg) ? __expf(s0[v] - mL[row]) * liL[row] : 0.0f;
      const float p1 = (sg0 + 16 < tg) ? __expf(s1[v] - mL[row]) * liL[row] : 0.0f;
      Pl[row * PP + ca + mrow]      = (_Float16)p0;
      Pl[row * PP + ca + mrow + 16] = (_Float16)p1;
    }
    __syncthreads();
    // A fragments of P (M=t, K=s in [0,64))
    v16h aP[4][2];
#pragma unroll
    for (int rt = 0; rt < 4; ++rt)
#pragma unroll
      for (int s2 = 0; s2 < 2; ++s2)
        aP[rt][s2] = ldsFragA(Pl, rt << 4, PP, s2 << 5, mrow, hi);
    // ---- Phase B: acc += P @ xh[sBase:sBase+64, :], K split in two 32s ----
#pragma unroll
    for (int s2 = 0; s2 < 2; ++s2) {
      __syncthreads();
      // async-stage XT[d=0..1023][k=0..31] = xhT[b][d][sBase + s2*32 + k]
#pragma unroll
      for (int rr = 0; rr < 16; ++rr) {
        const int q = tid + (rr << 8);
        const int row = q >> 2, c8 = (q & 3) << 3;
        asyncLoadB128(XT + row * TP + c8,
                      xhTb + (size_t)row * Tn + sBase + (s2 << 5) + c8);
      }
      asyncWait0();
      __syncthreads();
#pragma unroll
      for (int ct = 0; ct < 8; ++ct) {
        const v16h bf = ldsFragB(XT, (w << 7) + (ct << 4), TP, 0, mrow, hi);
#pragma unroll
        for (int rt = 0; rt < 4; ++rt)
          acc[rt][ct] = wmma_f16(aP[rt][s2], bf, acc[rt][ct]);
      }
    }
  }
  // ---- store out = -acc ----
#pragma unroll
  for (int rt = 0; rt < 4; ++rt)
#pragma unroll
    for (int ct = 0; ct < 8; ++ct)
#pragma unroll
      for (int v = 0; v < 8; ++v) {
        const int t = tBase + (rt << 4) + v + 8 * hi;
        const int d = (w << 7) + (ct << 4) + mrow;
        out[((size_t)b * Tn + t) * Cn + d] = -acc[rt][ct][v];
      }
}

// ============================================================================
// Host launch.  Workspace layout (needs ~64.1 MB):
//   xh   f16 [B][T][C]   33,554,432 B
//   xhT  f16 [B][C][T]   33,554,432 B
//   m    f32 [B][T]          65,536 B
//   l    f32 [B][T]          65,536 B
// ============================================================================
extern "C" void kernel_launch(void* const* d_in, const int* in_sizes, int n_in,
                              void* d_out, int out_size, void* d_ws,
                              size_t ws_size, hipStream_t stream) {
  (void)in_sizes; (void)n_in; (void)out_size; (void)ws_size;
  const float* x = (const float*)d_in[0];
  const float* W = (const float*)d_in[1];
  float* out = (float*)d_out;

  _Float16* xh  = (_Float16*)d_ws;
  _Float16* xhT = xh + (size_t)Bn * Tn * Cn;
  float* mbuf = (float*)(xhT + (size_t)Bn * Tn * Cn);
  float* lbuf = mbuf + (size_t)Bn * Tn;

  dim3 g1(Tn / 64, Cn / 64, Bn);   // 32 x 16 x 8
  k_gemm_xh<<<g1, 256, 0, stream>>>(x, W, xh, xhT);

  dim3 g2(Tn / 64, Bn);            // 32 x 8
  k_softmax_stats<<<g2, 256, 0, stream>>>(x, xh, mbuf, lbuf);
  k_attn_out<<<g2, 256, 0, stream>>>(x, xh, xhT, mbuf, lbuf, out);
}
